// GAT_11441792877056
// MI455X (gfx1250) — matrix operand order repaired and checked by reading, not compiled, not run on hardware
//
#include <hip/hip_runtime.h>

// ---------------------------------------------------------------------------
// GAT (3-layer, DGL-style) for MI455X / gfx1250.
//  - GEMMs: v_wmma_f32_16x16x32_bf16, fully unrolled K (templated), with W
//    pre-packed into the native B-fragment layout (contiguous b128 loads).
//  - Edge softmax + aggregation: L2-resident float atomics
//    (global_atomic_add_f32 / global_atomic_max_num_f32), float4 gathers.
// ---------------------------------------------------------------------------

#define NNODES 50000
#define NEDGES 1600000

typedef __bf16 bf16_t;
typedef __attribute__((ext_vector_type(16))) __bf16 v16bf;
typedef __attribute__((ext_vector_type(8)))  float v8f;

// ---------------------------------------------------------------------------
__global__ void init_f32(float* __restrict__ p, float v, long n) {
  long i = (long)blockIdx.x * blockDim.x + threadIdx.x;
  if (i < n) p[i] = v;
}

// ---------------------------------------------------------------------------
// Pack W[K,F] (f32, row-major) into bf16 B-fragment layout:
//   Wpk[((tile_n*(K/32) + k0/32)*32 + lane)*16 + i] = bf16(W[(k0+2*(i&8)+8*(lane>>4)+(i&7))*F + tile_n*16+(lane&15)])
__global__ void pack_W_bf16(const float* __restrict__ W, bf16_t* __restrict__ Wpk,
                            int K, int F) {
  int t = blockIdx.x * blockDim.x + threadIdx.x;  // over K*F
  if (t >= K * F) return;
  int i    = t & 15;
  int lane = (t >> 4) & 31;
  int blk  = t >> 9;                 // tile_n*(K/32) + k0idx
  int nk   = K / 32;
  int tile_n = blk / nk;
  int k      = (blk % nk) * 32 + 2 * (i & 8) + 8 * (lane >> 4) + (i & 7);
  int col    = tile_n * 16 + (lane & 15);
  Wpk[t] = (bf16_t)W[(long)k * F + col];
}

// ---------------------------------------------------------------------------
// C[M,F] = A[M,K] @ W[K,F].  One wave32 per 16x16 tile; K fully unrolled.
template <int K, int F>
__global__ void gemm_wmma_bf16(const float* __restrict__ A,
                               const bf16_t* __restrict__ Wpk,
                               float* __restrict__ C, int M) {
  const int lane   = threadIdx.x;                        // 0..31
  const int tile_n = blockIdx.x;                         // F/16 tiles
  const int tile_m = blockIdx.y * blockDim.y + threadIdx.y;
  if (tile_m * 16 >= M) return;                          // wave-uniform exit
  const int  half    = lane >> 4;                        // 0 or 1
  const int  l16     = lane & 15;
  const long rowbase = (long)(tile_m * 16 + l16) * K;    // A row for this lane

  v8f acc = {};
#pragma unroll
  for (int k0 = 0; k0 < K; k0 += 32) {
    // A fragment: lane holds K = [8h .. 8h+7] and [16+8h .. 23+8h]  (h = half)
    float abuf[16];
    *(float4*)&abuf[0]  = *(const float4*)(A + rowbase + k0 + 8 * half);
    *(float4*)&abuf[4]  = *(const float4*)(A + rowbase + k0 + 8 * half + 4);
    *(float4*)&abuf[8]  = *(const float4*)(A + rowbase + k0 + 16 + 8 * half);
    *(float4*)&abuf[12] = *(const float4*)(A + rowbase + k0 + 16 + 8 * half + 4);
    v16bf a;
#pragma unroll
    for (int i = 0; i < 16; ++i) a[i] = (bf16_t)abuf[i];
    // B fragment: contiguous 32B per lane from the packed layout
    v16bf b = *(const v16bf*)(Wpk +
        ((long)(tile_n * (K / 32) + (k0 >> 5)) * 32 + lane) * 16);
    acc = __builtin_amdgcn_wmma_f32_16x16x32_bf16(
        /*neg_a=*/false, a, /*neg_b=*/false, b,
        /*c_mod=*/(short)0, acc, /*reuse_a=*/false, /*reuse_b=*/false);
  }
  // D layout: VGPR r -> lanes 0-15: M=r, lanes 16-31: M=r+8; N = l16
#pragma unroll
  for (int r = 0; r < 8; ++r)
    C[(long)(tile_m * 16 + r + 8 * half) * F + tile_n * 16 + l16] = acc[r];
}

template <int K, int F>
static void launch_gemm(const float* A, const float* W, bf16_t* Wpk, float* C,
                        int M, hipStream_t stream) {
  int tot = K * F;
  pack_W_bf16<<<(tot + 255) / 256, 256, 0, stream>>>(W, Wpk, K, F);
  dim3 blk(32, 4);
  dim3 grd(F / 16, (M / 16 + 3) / 4);
  gemm_wmma_bf16<K, F><<<grd, blk, 0, stream>>>(A, Wpk, C, M);
}

// ---------------------------------------------------------------------------
// el[n,h] = sum_d feat[n,h,d]*al[h,d] ; er likewise with ar.
__global__ void attn_proj(const float* __restrict__ feat,
                          const float* __restrict__ al,
                          const float* __restrict__ ar,
                          float* __restrict__ el, float* __restrict__ er,
                          int M, int H, int D) {
  int t = blockIdx.x * blockDim.x + threadIdx.x;
  if (t >= M * H) return;
  int n = t / H, h = t % H;
  const float* fr = feat + (long)n * H * D + (long)h * D;
  float sl = 0.f, sr = 0.f;
  for (int d4 = 0; d4 < D; d4 += 4) {
    float4 v = *(const float4*)(fr + d4);
    float4 l = *(const float4*)(al + h * D + d4);
    float4 r = *(const float4*)(ar + h * D + d4);
    sl += v.x * l.x + v.y * l.y + v.z * l.z + v.w * l.w;
    sr += v.x * r.x + v.y * r.y + v.z * r.z + v.w * r.w;
  }
  el[t] = sl;
  er[t] = sr;
}

// ---------------------------------------------------------------------------
// eatt[e,h] = leaky_relu(el[src]+er[dst]); mx[dst,h] = segment max
__global__ void edge_logit_max(const int* __restrict__ src,
                               const int* __restrict__ dst,
                               const float* __restrict__ el,
                               const float* __restrict__ er,
                               float* __restrict__ eatt,
                               float* __restrict__ mx,
                               long EH, int H) {
  long t = (long)blockIdx.x * blockDim.x + threadIdx.x;
  if (t >= EH) return;
  long e = t / H;
  int  h = (int)(t % H);
  int  s = src[e], d = dst[e];
  float v = el[(long)s * H + h] + er[(long)d * H + h];
  v = v > 0.f ? v : 0.2f * v;                 // leaky relu, slope 0.2
  eatt[t] = v;
  atomicMax(&mx[(long)d * H + h], v);         // global_atomic_max_num_f32
}

// eatt[e,h] = exp(eatt - mx[dst]); den[dst,h] += eatt
__global__ void edge_exp_sum(const int* __restrict__ dst,
                             const float* __restrict__ mx,
                             float* __restrict__ eatt,
                             float* __restrict__ den,
                             long EH, int H) {
  long t = (long)blockIdx.x * blockDim.x + threadIdx.x;
  if (t >= EH) return;
  long e = t / H;
  int  h = (int)(t % H);
  int  d = dst[e];
  float ex = __expf(eatt[t] - mx[(long)d * H + h]);
  eatt[t] = ex;
  atomicAdd(&den[(long)d * H + h], ex);
}

// rst[dst, f..f+3] += (ex/den[dst]) * feat[src, f..f+3]
__global__ void edge_scatter4(const int* __restrict__ src,
                              const int* __restrict__ dst,
                              const float* __restrict__ ex,
                              const float* __restrict__ den,
                              const float* __restrict__ feat,
                              float* __restrict__ rst,
                              long total4, int F, int D, int H) {
  long t = (long)blockIdx.x * blockDim.x + threadIdx.x;
  if (t >= total4) return;
  const int q4 = F >> 2;
  long e = t / q4;
  int  q = (int)(t % q4);
  int  f = q << 2;
  int  h = f / D;
  int  s = src[e], d = dst[e];
  float alpha = ex[e * H + h] / den[(long)d * H + h];
  float4 fv = *(const float4*)(feat + (long)s * F + f);
  float* rp = rst + (long)d * F + f;
  atomicAdd(rp + 0, alpha * fv.x);
  atomicAdd(rp + 1, alpha * fv.y);
  atomicAdd(rp + 2, alpha * fv.z);
  atomicAdd(rp + 3, alpha * fv.w);
}

// out = elu(rst + bias [+ resid])
__global__ void finalize_elu(const float* __restrict__ rst,
                             const float* __restrict__ resid,  // may be null
                             const float* __restrict__ bias,
                             float* __restrict__ out,
                             long total, int F) {
  long t = (long)blockIdx.x * blockDim.x + threadIdx.x;
  if (t >= total) return;
  int f = (int)(t % F);
  float v = rst[t] + bias[f];
  if (resid) v += resid[t];
  out[t] = v > 0.f ? v : expm1f(v);
}

// out[n,o] = mean_h( rst[n,h,o] + res[n,h,o] + bias[h,o] )
__global__ void finalize_mean(const float* __restrict__ rst,
                              const float* __restrict__ res,
                              const float* __restrict__ bias,
                              float* __restrict__ out,
                              int M, int H, int D) {
  int t = blockIdx.x * blockDim.x + threadIdx.x;
  if (t >= M * D) return;
  int n = t / D, o = t % D;
  float s = 0.f;
  for (int h = 0; h < H; ++h) {
    long idx = (long)n * H * D + (long)h * D + o;
    s += rst[idx] + res[idx] + bias[h * D + o];
  }
  out[(long)n * D + o] = s * (1.0f / H);
}

// ---------------------------------------------------------------------------
extern "C" void kernel_launch(void* const* d_in, const int* in_sizes, int n_in,
                              void* d_out, int out_size, void* d_ws, size_t ws_size,
                              hipStream_t stream) {
  const float* x    = (const float*)d_in[0];
  const int*   src  = (const int*)  d_in[1];
  const int*   dst  = (const int*)  d_in[2];
  const float* W1   = (const float*)d_in[3];
  const float* al1  = (const float*)d_in[4];
  const float* ar1  = (const float*)d_in[5];
  const float* b1   = (const float*)d_in[6];
  const float* W2   = (const float*)d_in[7];
  const float* al2  = (const float*)d_in[8];
  const float* ar2  = (const float*)d_in[9];
  const float* b2   = (const float*)d_in[10];
  const float* W3   = (const float*)d_in[11];
  const float* al3  = (const float*)d_in[12];
  const float* ar3  = (const float*)d_in[13];
  const float* b3   = (const float*)d_in[14];
  const float* rW3  = (const float*)d_in[15];
  float* out = (float*)d_out;
  (void)in_sizes; (void)n_in; (void)out_size; (void)ws_size;

  // ---- workspace carve-out (floats; all sizes multiples of 8 -> 32B align) ----
  float* ws = (float*)d_ws;
  size_t off = 0;
  auto carve = [&](size_t n) { float* p = ws + off; off += n; return p; };
  float* h1   = carve((size_t)NNODES * 128);
  float* h2   = carve((size_t)NNODES * 128);
  float* feat = carve((size_t)NNODES * 192);
  float* rst  = carve((size_t)NNODES * 192);
  float* res3 = carve((size_t)NNODES * 192);
  float* el   = carve((size_t)NNODES * 6);
  float* er   = carve((size_t)NNODES * 6);
  float* mx   = carve((size_t)NNODES * 6);
  float* den  = carve((size_t)NNODES * 6);
  float* eatt = carve((size_t)NEDGES * 6);
  bf16_t* Wpk = (bf16_t*)carve(128 * 192 / 2);  // packed weights (bf16)

  // edge softmax + weighted aggregation (feat -> rst), H heads of dim 32
  auto softmax_agg = [&](int H) {
    const int F = H * 32;
    int nh = NNODES * H;
    attn_proj<<<(nh + 255) / 256, 256, 0, stream>>>(feat,
        (H == 4) ? nullptr : nullptr, nullptr, el, er, NNODES, H, 32);
    (void)0;
  };
  (void)softmax_agg;  // (expanded inline below with proper al/ar pointers)

  auto run_edges = [&](const float* al, const float* ar, int H) {
    const int F = H * 32;
    int nh = NNODES * H;
    attn_proj<<<(nh + 255) / 256, 256, 0, stream>>>(feat, al, ar, el, er,
                                                    NNODES, H, 32);
    long nH = (long)NNODES * H;
    long nF = (long)NNODES * F;
    init_f32<<<(unsigned)((nH + 255) / 256), 256, 0, stream>>>(mx, -3.0e38f, nH);
    init_f32<<<(unsigned)((nH + 255) / 256), 256, 0, stream>>>(den, 0.f, nH);
    init_f32<<<(unsigned)((nF + 255) / 256), 256, 0, stream>>>(rst, 0.f, nF);
    long eh = (long)NEDGES * H;
    edge_logit_max<<<(unsigned)((eh + 255) / 256), 256, 0, stream>>>(
        src, dst, el, er, eatt, mx, eh, H);
    edge_exp_sum<<<(unsigned)((eh + 255) / 256), 256, 0, stream>>>(
        dst, mx, eatt, den, eh, H);
    long ef4 = (long)NEDGES * (F >> 2);
    edge_scatter4<<<(unsigned)((ef4 + 255) / 256), 256, 0, stream>>>(
        src, dst, eatt, den, feat, rst, ef4, F, 32, H);
  };

  // ---- layer 1: 128 -> (4,32), no residual, ELU ----
  launch_gemm<128, 128>(x, W1, Wpk, feat, NNODES, stream);
  run_edges(al1, ar1, 4);
  {
    long tot = (long)NNODES * 128;
    finalize_elu<<<(unsigned)((tot + 255) / 256), 256, 0, stream>>>(
        rst, nullptr, b1, h1, tot, 128);
  }

  // ---- layer 2: 128 -> (4,32), identity residual, ELU ----
  launch_gemm<128, 128>(h1, W2, Wpk, feat, NNODES, stream);
  run_edges(al2, ar2, 4);
  {
    long tot = (long)NNODES * 128;
    finalize_elu<<<(unsigned)((tot + 255) / 256), 256, 0, stream>>>(
        rst, h1, b2, h2, tot, 128);
  }

  // ---- layer 3: 128 -> (6,32), linear residual, no act, head-mean ----
  launch_gemm<128, 192>(h2, W3, Wpk, feat, NNODES, stream);
  run_edges(al3, ar3, 6);
  launch_gemm<128, 192>(h2, rW3, Wpk, res3, NNODES, stream);
  {
    int tot = NNODES * 32;
    finalize_mean<<<(tot + 255) / 256, 256, 0, stream>>>(
        rst, res3, b3, out, NNODES, 6, 32);
  }
}
